// GraphNetwork_kMVN_53996328845316
// MI455X (gfx1250) — compile-verified, matrix operand order checked by reading, not run on hardware
//
#include <hip/hip_runtime.h>
#include <math.h>

// ---------------------------------------------------------------------------
// MI455X (gfx1250, wave32) implementation.
//
// Node-side einsums  einsum('nu,nv,uvk->nk')  are executed as bf16 WMMA GEMMs:
//   A (16 nodes x 32 red.)  = outer-product rows  x[row,kb] * z[row,:]
//   B (32 red. x 32 out)    = w[u=kb, v, k]  pre-swizzled to WMMA lane order
//   D accumulated in f32 via v_wmma_f32_16x16x32_bf16.
// B-fragment loads are software-pipelined one K-step ahead so WMMAs do not
// sit behind s_wait_loadcnt 0x0 (overlap L2 latency with the matrix pipe).
// Edge phase: one wave per edge; MLP weights LDS-resident; scatter with
// global f32 atomics (accumulators are L2-resident: 57.6MB << 192MB L2).
// ---------------------------------------------------------------------------

typedef __attribute__((ext_vector_type(16))) __bf16 v16bf;
typedef __attribute__((ext_vector_type(8)))  float  v8f;

#define WMMA_BF16(a, b, c) \
  __builtin_amdgcn_wmma_f32_16x16x32_bf16(false, (a), false, (b), (short)0, (c), false, false)

// 16-bit A/B matrix lane layout (CDNA5 ISA 7.12.2):
//  lanes 0-15 : K elements {0..7} in regs 0-3, {16..23} in regs 4-7
//  lanes 16-31: K elements {8..15},          {24..31}
__device__ __forceinline__ int kmap(int lane, int i) {
  return ((i & 8) << 1) | ((lane >> 4) << 3) | (i & 7);
}

static constexpr float INV_CA = 1.0f / 32.0f;      // 1/sqrt(32*32)
static constexpr float INV_SB = 0.31622776601f;    // 1/sqrt(10)
static constexpr float INV_SH = 0.125f;            // 1/sqrt(64)
static constexpr float C_S = 0.38268343236f;       // sin(pi/8)
static constexpr float C_X = 0.92387953251f;       // cos(pi/8)

// ---------------------------------------------------------------------------
// Kernel 0: swizzle the five (32,32,32) f32 weight tensors into bf16 WMMA
// B-fragment order: flat index = (((kb*2 + ct)*32 + lane)*16 + i)
// source element   = w[u=kb, v=kmap(lane,i), k=ct*16 + (lane&15)]
// ---------------------------------------------------------------------------
__global__ void swizzle_w_kernel(const float* __restrict__ w0,
                                 const float* __restrict__ w1,
                                 const float* __restrict__ w2,
                                 const float* __restrict__ w3,
                                 const float* __restrict__ w4,
                                 __bf16* __restrict__ wswz) {
  int idx = blockIdx.x * blockDim.x + threadIdx.x;
  if (idx >= 5 * 32768) return;
  int t = idx >> 15;
  int r = idx & 32767;
  int i    = r & 15;
  int lane = (r >> 4) & 31;
  int ct   = (r >> 9) & 1;
  int kb   = r >> 10;
  int v = kmap(lane, i);
  int n = ct * 16 + (lane & 15);
  const float* src = (t == 0) ? w0 : (t == 1) ? w1 : (t == 2) ? w2 : (t == 3) ? w3 : w4;
  wswz[idx] = (__bf16)src[kb * 1024 + v * 32 + n];
}

// ---------------------------------------------------------------------------
// Kernel 1: per-node pre tensor products -> nf = h0*inv_ca*deg^-1/2, mask.
// One wave per 16-node tile; 2 tensors x 2 col-tiles x 32 steps = 128 WMMA,
// with B fragments double-buffered one step ahead.
// ---------------------------------------------------------------------------
__global__ void node_pre_kernel(const float* __restrict__ x,     // node_input  N x 32
                                const float* __restrict__ z,     // node_attr   N x 32
                                const float* __restrict__ deg,   // node_deg    N
                                const __bf16* __restrict__ wIn,  // swizzled w_lin_in
                                const __bf16* __restrict__ wMk,  // swizzled w_lin_mask
                                float* __restrict__ nf,
                                float* __restrict__ mask,
                                int N, int nTiles) {
  const int tile = blockIdx.x * (blockDim.x >> 5) + (threadIdx.x >> 5);
  if (tile >= nTiles) return;                       // wave-uniform: EXEC stays all-1s
  const int lane = threadIdx.x & 31;
  const int row  = lane & 15;
  const int node = tile * 16 + row;
  const int nodec = node < N ? node : N - 1;

  __builtin_prefetch(wIn, 0, 1);                    // global_prefetch_b8
  __builtin_prefetch(wMk, 0, 1);

  float zf[16];
#pragma unroll
  for (int i = 0; i < 16; ++i) zf[i] = z[nodec * 32 + kmap(lane, i)];

  v8f acc0 = {}, acc1 = {}, acc2 = {}, acc3 = {};
  const v16bf* fA = (const v16bf*)wIn;
  const v16bf* fB = (const v16bf*)wMk;

  // prime the pipeline with kb=0 fragments
  v16bf b0 = fA[lane];
  v16bf b1 = fA[32 + lane];
  v16bf b2 = fB[lane];
  v16bf b3 = fB[32 + lane];

#pragma unroll
  for (int kb = 0; kb < 32; ++kb) {
    const int kn = (kb + 1) & 31;                   // wrap: no branch, EXEC uniform
    v16bf n0 = fA[(kn * 2 + 0) * 32 + lane];
    v16bf n1 = fA[(kn * 2 + 1) * 32 + lane];
    v16bf n2 = fB[(kn * 2 + 0) * 32 + lane];
    v16bf n3 = fB[(kn * 2 + 1) * 32 + lane];

    float xk = x[nodec * 32 + kb];
    v16bf a;
#pragma unroll
    for (int i = 0; i < 16; ++i) a[i] = (__bf16)(zf[i] * xk);

    acc0 = WMMA_BF16(a, b0, acc0);
    acc1 = WMMA_BF16(a, b1, acc1);
    acc2 = WMMA_BF16(a, b2, acc2);
    acc3 = WMMA_BF16(a, b3, acc3);

    b0 = n0; b1 = n1; b2 = n2; b3 = n3;
  }

  // D layout: reg j -> M = j + 8*(lane>=16), N = lane&15
  const int mb = (lane >> 4) * 8;
  const int nc = lane & 15;
#pragma unroll
  for (int j = 0; j < 8; ++j) {
    int nrow = tile * 16 + mb + j;
    if (nrow < N) {
      float dis = rsqrtf(deg[nrow]);
      nf[nrow * 32 + nc]        = acc0[j] * INV_CA * dis;
      nf[nrow * 32 + 16 + nc]   = acc1[j] * INV_CA * dis;
      mask[nrow * 32 + nc]      = acc2[j] * INV_CA;
      mask[nrow * 32 + 16 + nc] = acc3[j] * INV_CA;
    }
  }
}

// ---------------------------------------------------------------------------
// Kernel 2: edge phase. One wave per edge. W0 (10x64) and W1 (64x96) live in
// LDS; each lane owns channel c = lane and issues 9 f32 atomics (L2-resident).
// ---------------------------------------------------------------------------
__global__ void edge_kernel(const int*   __restrict__ esrc,
                            const int*   __restrict__ edst,
                            const float* __restrict__ eattr,   // E x 9
                            const float* __restrict__ el,      // E x 10
                            const float* __restrict__ W0,      // 10 x 64
                            const float* __restrict__ W1,      // 64 x 96
                            const float* __restrict__ nf,      // N x 32
                            float* __restrict__ a0,            // N x 32
                            float* __restrict__ a1,            // N x 96
                            float* __restrict__ a2,            // N x 160
                            int E) {
  __shared__ float sW0[10 * 64];
  __shared__ float sW1[64 * 96];
  __shared__ float sEl[8][10];
  __shared__ float sAttr[8][9];
  __shared__ float sH[8][64];

  for (int i = threadIdx.x; i < 640; i += blockDim.x) sW0[i] = W0[i];
  for (int i = threadIdx.x; i < 6144; i += blockDim.x) sW1[i] = W1[i];
  __syncthreads();

  const int w = threadIdx.x >> 5;
  const int lane = threadIdx.x & 31;
  const int e = blockIdx.x * 8 + w;
  if (e >= E) return;                                // wave-uniform

  if (lane < 10) sEl[w][lane] = el[e * 10 + lane];
  if (lane < 9)  sAttr[w][lane] = eattr[e * 9 + lane];
  // same-wave LDS RAW: compiler inserts s_wait_dscnt

  // h = silu(el @ W0 / sqrt(B)) : lane computes h[lane], h[lane+32]
#pragma unroll
  for (int t = 0; t < 2; ++t) {
    int j = lane + t * 32;
    float s = 0.f;
#pragma unroll
    for (int b = 0; b < 10; ++b) s += sEl[w][b] * sW0[b * 64 + j];
    s *= INV_SB;
    sH[w][j] = s / (1.0f + __expf(-s));
  }

  // ew = h @ W1 / sqrt(H) : lane computes ew[lane], ew[32+lane], ew[64+lane]
  float ew0 = 0.f, ew1 = 0.f, ew2 = 0.f;
#pragma unroll 4
  for (int j = 0; j < 64; ++j) {
    float h = sH[w][j];
    ew0 += h * sW1[j * 96 + lane];
    ew1 += h * sW1[j * 96 + 32 + lane];
    ew2 += h * sW1[j * 96 + 64 + lane];
  }
  ew0 *= INV_SH; ew1 *= INV_SH; ew2 *= INV_SH;

  const int src = esrc[e];
  const int dst = edst[e];
  const float xs = nf[src * 32 + lane];

  float m0 = ew0 * xs * sAttr[w][0];
  __hip_atomic_fetch_add(&a0[dst * 32 + lane], m0, __ATOMIC_RELAXED, __HIP_MEMORY_SCOPE_AGENT);

  const float w1x = ew1 * xs;
#pragma unroll
  for (int d = 0; d < 3; ++d)
    __hip_atomic_fetch_add(&a1[dst * 96 + lane * 3 + d], w1x * sAttr[w][1 + d],
                           __ATOMIC_RELAXED, __HIP_MEMORY_SCOPE_AGENT);

  const float w2x = ew2 * xs;
#pragma unroll
  for (int q = 0; q < 5; ++q)
    __hip_atomic_fetch_add(&a2[dst * 160 + lane * 5 + q], w2x * sAttr[w][4 + q],
                           __ATOMIC_RELAXED, __HIP_MEMORY_SCOPE_AGENT);
}

// ---------------------------------------------------------------------------
// Kernel 3: per-node post tensor products. One wave per 16-node tile;
// 9 GEMM slots (out0, 3x out1, 5x out2) x 2 col-tiles x 32 steps = 576 WMMA,
// B fragments double-buffered one K-step ahead.
// Fuses deg^-1/2 scaling, inv_ca, mask mix, and the 288-wide output layout.
// ---------------------------------------------------------------------------
__global__ void node_post_kernel(const float* __restrict__ z,     // node_attr
                                 const float* __restrict__ deg,
                                 const __bf16* __restrict__ w0s,  // swizzled w_out0
                                 const __bf16* __restrict__ w1s,  // swizzled w_out1
                                 const __bf16* __restrict__ w2s,  // swizzled w_out2
                                 const float* __restrict__ a0,
                                 const float* __restrict__ a1,
                                 const float* __restrict__ a2,
                                 const float* __restrict__ mask,
                                 float* __restrict__ out,         // N x 288
                                 int N, int nTiles) {
  const int tile = blockIdx.x * (blockDim.x >> 5) + (threadIdx.x >> 5);
  if (tile >= nTiles) return;                        // wave-uniform
  const int lane = threadIdx.x & 31;
  const int row  = lane & 15;
  const int node = tile * 16 + row;
  const int nodec = node < N ? node : N - 1;

  __builtin_prefetch(w0s, 0, 1);
  __builtin_prefetch(w1s, 0, 1);
  __builtin_prefetch(w2s, 0, 1);

  float zf[16];
#pragma unroll
  for (int i = 0; i < 16; ++i) zf[i] = z[nodec * 32 + kmap(lane, i)];
  const float dis = rsqrtf(deg[nodec]);

  const int mb = (lane >> 4) * 8;
  const int nc = lane & 15;

  for (int s = 0; s < 9; ++s) {
    const __bf16* wsel = (s == 0) ? w0s : (s < 4) ? w1s : w2s;
    const v16bf* fw = (const v16bf*)wsel;

    float xv[32];
    if (s == 0) {
#pragma unroll
      for (int u = 0; u < 32; ++u) xv[u] = a0[nodec * 32 + u] * dis;
    } else if (s < 4) {
      int d = s - 1;
#pragma unroll
      for (int u = 0; u < 32; ++u) xv[u] = a1[nodec * 96 + u * 3 + d] * dis;
    } else {
      int q = s - 4;
#pragma unroll
      for (int u = 0; u < 32; ++u) xv[u] = a2[nodec * 160 + u * 5 + q] * dis;
    }

    v8f acc0 = {}, acc1 = {};
    v16bf b0 = fw[lane];                    // kb=0 fragments
    v16bf b1 = fw[32 + lane];

#pragma unroll
    for (int kb = 0; kb < 32; ++kb) {
      const int kn = (kb + 1) & 31;         // wrap: branch-free pipeline
      v16bf n0 = fw[(kn * 2 + 0) * 32 + lane];
      v16bf n1 = fw[(kn * 2 + 1) * 32 + lane];

      float xk = xv[kb];
      v16bf a;
#pragma unroll
      for (int i = 0; i < 16; ++i) a[i] = (__bf16)(zf[i] * xk);

      acc0 = WMMA_BF16(a, b0, acc0);
      acc1 = WMMA_BF16(a, b1, acc1);

      b0 = n0; b1 = n1;
    }

#pragma unroll
    for (int j = 0; j < 8; ++j) {
      int nrow = tile * 16 + mb + j;
      if (nrow < N) {
        float v0 = acc0[j] * INV_CA;
        float v1 = acc1[j] * INV_CA;
        if (s == 0) {
          out[nrow * 288 + nc]      = C_S * mask[nrow * 32 + nc]      + C_X * v0;
          out[nrow * 288 + 16 + nc] = C_S * mask[nrow * 32 + 16 + nc] + C_X * v1;
        } else if (s < 4) {
          int d = s - 1;
          out[nrow * 288 + 32 + nc * 3 + d]        = v0;   // out1[n,k,m], k=nc
          out[nrow * 288 + 32 + (16 + nc) * 3 + d] = v1;
        } else {
          int q = s - 4;
          out[nrow * 288 + 128 + nc * 5 + q]        = v0;  // out2[n,k,m]
          out[nrow * 288 + 128 + (16 + nc) * 5 + q] = v1;
        }
      }
    }
  }
}

// ---------------------------------------------------------------------------
extern "C" void kernel_launch(void* const* d_in, const int* in_sizes, int n_in,
                              void* d_out, int out_size, void* d_ws, size_t ws_size,
                              hipStream_t stream) {
  const float* node_input = (const float*)d_in[0];
  const float* node_attr  = (const float*)d_in[1];
  const float* node_deg   = (const float*)d_in[2];
  const int*   edge_src   = (const int*)d_in[3];
  const int*   edge_dst   = (const int*)d_in[4];
  const float* edge_attr  = (const float*)d_in[5];
  const float* edge_len   = (const float*)d_in[6];
  const float* w_lin_in   = (const float*)d_in[7];
  const float* w_lin_mask = (const float*)d_in[8];
  const float* w_out0     = (const float*)d_in[9];
  const float* w_out1     = (const float*)d_in[10];
  const float* w_out2     = (const float*)d_in[11];
  const float* W0         = (const float*)d_in[12];
  const float* W1         = (const float*)d_in[13];
  // d_in[14] (numb) and d_in[15] (n) are unused by the reference body.

  const int N = in_sizes[0] / 32;
  const int E = in_sizes[3];
  const int nTiles = (N + 15) / 16;

  // Workspace layout (bytes). Total ~ 70.8 MB for N=50000.
  char* ws = (char*)d_ws;
  const size_t off_w    = 0;                                  // 5*32768 bf16
  const size_t off_nf   = off_w + 5 * 32768 * sizeof(__bf16); // N*32 f32
  const size_t off_mask = off_nf   + (size_t)N * 32 * 4;
  const size_t off_a0   = off_mask + (size_t)N * 32 * 4;
  const size_t off_a1   = off_a0   + (size_t)N * 32 * 4;
  const size_t off_a2   = off_a1   + (size_t)N * 96 * 4;
  (void)ws_size;

  __bf16* wswz = (__bf16*)(ws + off_w);
  float* nf    = (float*)(ws + off_nf);
  float* mask  = (float*)(ws + off_mask);
  float* a0    = (float*)(ws + off_a0);
  float* a1    = (float*)(ws + off_a1);
  float* a2    = (float*)(ws + off_a2);

  // Zero the edge accumulators (a0|a1|a2 are contiguous: N*288 floats).
  hipMemsetAsync(a0, 0, (size_t)N * 288 * 4, stream);

  // 0) swizzle weights to bf16 WMMA fragment order
  swizzle_w_kernel<<<(5 * 32768 + 255) / 256, 256, 0, stream>>>(
      w_lin_in, w_lin_mask, w_out0, w_out1, w_out2, wswz);

  // 1) node pre tensor products (WMMA)
  const int preBlocks = (nTiles + 3) / 4;  // 4 waves / 128-thread block
  node_pre_kernel<<<preBlocks, 128, 0, stream>>>(
      node_input, node_attr, node_deg,
      wswz + 0 * 32768, wswz + 1 * 32768, nf, mask, N, nTiles);

  // 2) edge MLP + gather + atomic scatter (one wave per edge)
  edge_kernel<<<(E + 7) / 8, 256, 0, stream>>>(
      edge_src, edge_dst, edge_attr, edge_len, W0, W1, nf, a0, a1, a2, E);

  // 3) node post tensor products (WMMA) + output assembly
  node_post_kernel<<<preBlocks, 128, 0, stream>>>(
      node_attr, node_deg,
      wswz + 2 * 32768, wswz + 3 * 32768, wswz + 4 * 32768,
      a0, a1, a2, mask, (float*)d_out, N, nTiles);
}
